// ResNHConv_274877907666
// MI455X (gfx1250) — matrix operand order repaired and checked by reading, not compiled, not run on hardware
//
#include <hip/hip_runtime.h>
#include <hip/hip_bf16.h>
#include <math.h>

typedef __attribute__((ext_vector_type(16))) _Float16 v16h;
typedef __attribute__((ext_vector_type(8)))  _Float16 v8h;
typedef __attribute__((ext_vector_type(8)))  float    v8f;

#define FDIM 64
#define NHEAD 9
#define LN_EPS 1e-5f

__device__ __forceinline__ float silu_f(float x) { return x / (1.0f + __expf(-x)); }

// ---------------------------------------------------------------------------
// Pack W1/W2 (f32, [9][64][64]) into the WMMA B-operand lane layout, f16.
// Layout: wp[((((layer*9+k)*2+c)*4+t)*32 + lane)*16 + e]
//   B tile (32x16): col N = t*16 + (lane&15), K = c*32 + (lane>>4)*8 + (e&7) + (e>>3)*16
// ---------------------------------------------------------------------------
__global__ void pack_w(const float* __restrict__ W1, const float* __restrict__ W2,
                       _Float16* __restrict__ wp) {
  int tid = blockIdx.x * 256 + threadIdx.x;       // 0 .. 73727
  int layer = tid / 36864;
  int r = tid % 36864;
  int k = r / 4096;  r %= 4096;
  int c = r / 2048;  r %= 2048;
  int t = r / 512;   r %= 512;
  int lane = r / 16;
  int e = r % 16;
  int K = c * 32 + (lane >> 4) * 8 + (e & 7) + (e >> 3) * 16;
  int N = t * 16 + (lane & 15);
  const float* W = layer ? W2 : W1;
  wp[tid] = (_Float16)W[(k * FDIM + K) * FDIM + N];
}

// ---------------------------------------------------------------------------
// t1[n][f] = f16( silu( layernorm(x[n]) * w + b ) )
// One wave per 16 rows: lane handles row (lane&15), half (lane>>4) covers
// 32 features; cross-half combine via shfl_xor(16) (wave32).
// ---------------------------------------------------------------------------
__global__ void ln_silu(const float* __restrict__ x, const float* __restrict__ w,
                        const float* __restrict__ b, _Float16* __restrict__ t1) {
  int lane = threadIdx.x & 31;
  int wave = threadIdx.x >> 5;
  int row  = (blockIdx.x * 8 + wave) * 16 + (lane & 15);
  int hh   = lane >> 4;

  const float4* xr = (const float4*)(x + (size_t)row * FDIM + hh * 32);
  float v[32];
#pragma unroll
  for (int i = 0; i < 8; i++) {
    float4 q = xr[i];
    v[4*i] = q.x; v[4*i+1] = q.y; v[4*i+2] = q.z; v[4*i+3] = q.w;
  }
  float s = 0.f, ss = 0.f;
#pragma unroll
  for (int i = 0; i < 32; i++) { s += v[i]; ss += v[i] * v[i]; }
  s  += __shfl_xor(s, 16, 32);
  ss += __shfl_xor(ss, 16, 32);
  float mu  = s * (1.0f / FDIM);
  float inv = rsqrtf(ss * (1.0f / FDIM) - mu * mu + LN_EPS);

  const float4* wr = (const float4*)(w + hh * 32);
  const float4* br = (const float4*)(b + hh * 32);
  union { _Float16 h[32]; v8h v[4]; } o;
#pragma unroll
  for (int i = 0; i < 8; i++) {
    float4 wq = wr[i], bq = br[i];
    o.h[4*i]   = (_Float16)silu_f((v[4*i]   - mu) * inv * wq.x + bq.x);
    o.h[4*i+1] = (_Float16)silu_f((v[4*i+1] - mu) * inv * wq.y + bq.y);
    o.h[4*i+2] = (_Float16)silu_f((v[4*i+2] - mu) * inv * wq.z + bq.z);
    o.h[4*i+3] = (_Float16)silu_f((v[4*i+3] - mu) * inv * wq.w + bq.w);
  }
  v8h* dst = (v8h*)(t1 + (size_t)row * FDIM + hh * 32);
#pragma unroll
  for (int i = 0; i < 4; i++) dst[i] = o.v[i];
}

// ---------------------------------------------------------------------------
// One wave computes a 16-node x 64-feature output tile:
//   C[16x64] = sum_k gather(tin, adjc[:,k])[16x64] @ Wk[64x64]
// via v_wmma_f32_16x16x32_f16 (9 heads x 2 k-chunks x 4 n-tiles = 72 WMMA).
// FIRST: epilogue = +b1, LN2, SiLU -> f16 tout
// !FIRST: epilogue = +b2 + residual x -> f32 fout
// ---------------------------------------------------------------------------
template <bool FIRST>
__global__ void gemm_layer(const _Float16* __restrict__ tin,
                           const long long* __restrict__ adjc,
                           const _Float16* __restrict__ wp,   // layer-offset pre-applied
                           const float* __restrict__ bias,
                           const float* __restrict__ lnw,
                           const float* __restrict__ lnb,
                           const float* __restrict__ xres,
                           _Float16* __restrict__ tout,
                           float* __restrict__ fout) {
  int lane = threadIdx.x & 31;
  int wave = threadIdx.x >> 5;
  int base = (blockIdx.x * 8 + wave) * 16;
  int m  = lane & 15;          // A row / C col
  int hh = lane >> 4;          // half-wave
  int node = base + m;

  v8f acc[4] = {};

  const long long* arow = adjc + (size_t)node * NHEAD;
#pragma unroll
  for (int k = 0; k < NHEAD; k++) {
    long long nbr = arow[k];
    const _Float16* rowp = tin + (size_t)nbr * FDIM;
    // A operand, ISA 16-bit 16x32 layout:
    //   lane<16: e0-7 -> K 0-7,  e8-15 -> K 16-23
    //   lane>=16: e0-7 -> K 8-15, e8-15 -> K 24-31   (+32 for chunk c=1)
    v8h lo0 = *(const v8h*)(rowp + hh * 8);
    v8h hi0 = *(const v8h*)(rowp + hh * 8 + 16);
    v8h lo1 = *(const v8h*)(rowp + 32 + hh * 8);
    v8h hi1 = *(const v8h*)(rowp + 32 + hh * 8 + 16);
    v16h a0, a1;
#pragma unroll
    for (int j = 0; j < 8; j++) {
      a0[j] = lo0[j]; a0[j + 8] = hi0[j];
      a1[j] = lo1[j]; a1[j + 8] = hi1[j];
    }
    const _Float16* bp = wp + (size_t)k * 8 * 512 + (size_t)lane * 16;
#pragma unroll
    for (int t = 0; t < 4; t++) {
      v16h b0 = *(const v16h*)(bp + t * 512);          // c = 0
      acc[t] = __builtin_amdgcn_wmma_f32_16x16x32_f16(
          false, a0, false, b0, (short)0, acc[t], false, false);
      v16h b1 = *(const v16h*)(bp + 2048 + t * 512);   // c = 1
      acc[t] = __builtin_amdgcn_wmma_f32_16x16x32_f16(
          false, a1, false, b1, (short)0, acc[t], false, false);
    }
  }

  // C layout: VGPR r holds row M = r + 8*hh, col N = m  (tile t -> feat t*16+m)
  float bv[4];
#pragma unroll
  for (int t = 0; t < 4; t++) bv[t] = bias[t * 16 + m];

  if constexpr (FIRST) {
    float wv[4], lb[4];
#pragma unroll
    for (int t = 0; t < 4; t++) { wv[t] = lnw[t * 16 + m]; lb[t] = lnb[t * 16 + m]; }
#pragma unroll
    for (int r = 0; r < 8; r++) {
      float v0 = acc[0][r] + bv[0];
      float v1 = acc[1][r] + bv[1];
      float v2 = acc[2][r] + bv[2];
      float v3 = acc[3][r] + bv[3];
      float s  = v0 + v1 + v2 + v3;
      float ss = v0 * v0 + v1 * v1 + v2 * v2 + v3 * v3;
#pragma unroll
      for (int mask = 1; mask < 16; mask <<= 1) {   // reduce within 16-lane half
        s  += __shfl_xor(s,  mask, 32);
        ss += __shfl_xor(ss, mask, 32);
      }
      float mu  = s * (1.0f / FDIM);
      float inv = rsqrtf(ss * (1.0f / FDIM) - mu * mu + LN_EPS);
      size_t ob = (size_t)(base + r + hh * 8) * FDIM;
      tout[ob +  0 + m] = (_Float16)silu_f((v0 - mu) * inv * wv[0] + lb[0]);
      tout[ob + 16 + m] = (_Float16)silu_f((v1 - mu) * inv * wv[1] + lb[1]);
      tout[ob + 32 + m] = (_Float16)silu_f((v2 - mu) * inv * wv[2] + lb[2]);
      tout[ob + 48 + m] = (_Float16)silu_f((v3 - mu) * inv * wv[3] + lb[3]);
    }
  } else {
#pragma unroll
    for (int r = 0; r < 8; r++) {
      size_t ob = (size_t)(base + r + hh * 8) * FDIM;
#pragma unroll
      for (int t = 0; t < 4; t++) {
        size_t idx = ob + t * 16 + m;
        fout[idx] = acc[t][r] + bv[t] + xres[idx];
      }
    }
  }
}

// ---------------------------------------------------------------------------
extern "C" void kernel_launch(void* const* d_in, const int* in_sizes, int n_in,
                              void* d_out, int out_size, void* d_ws, size_t ws_size,
                              hipStream_t stream) {
  const float*     x    = (const float*)d_in[0];
  const long long* adjc = (const long long*)d_in[1];
  const float*     ln1w = (const float*)d_in[2];
  const float*     ln1b = (const float*)d_in[3];
  const float*     W1   = (const float*)d_in[4];
  const float*     b1   = (const float*)d_in[5];
  const float*     ln2w = (const float*)d_in[6];
  const float*     ln2b = (const float*)d_in[7];
  const float*     W2   = (const float*)d_in[8];
  const float*     b2   = (const float*)d_in[9];
  float* out = (float*)d_out;

  int N = in_sizes[0] / FDIM;   // 196608

  // workspace layout: [packed W: 2*36864 f16][t1: N*64 f16][t2: N*64 f16]
  char* ws = (char*)d_ws;
  _Float16* wp = (_Float16*)ws;
  _Float16* t1 = (_Float16*)(ws + 147456);
  _Float16* t2 = (_Float16*)(ws + 147456 + (size_t)N * FDIM * sizeof(_Float16));

  int groups = N / 128;         // 8 waves/block, 16 nodes/wave

  pack_w<<<288, 256, 0, stream>>>(W1, W2, wp);
  ln_silu<<<groups, 256, 0, stream>>>(x, ln1w, ln1b, t1);
  gemm_layer<true><<<groups, 256, 0, stream>>>(t1, adjc, wp, b1, ln2w, ln2b,
                                               nullptr, t2, nullptr);
  gemm_layer<false><<<groups, 256, 0, stream>>>(t2, adjc, wp + 36864, b2, nullptr,
                                                nullptr, x, nullptr, out);
}
